// LSTM_86698209837562
// MI455X (gfx1250) — 16-bit and 8-bit matrix operand order repaired and checked by reading, compiled, not run on hardware
//
#include <hip/hip_runtime.h>


typedef unsigned short u16;
typedef unsigned char  u8;
typedef __bf16        v16bf __attribute__((ext_vector_type(16)));
typedef float         v8f   __attribute__((ext_vector_type(8)));
typedef int           v8i   __attribute__((ext_vector_type(8)));
typedef unsigned int  u32x4 __attribute__((ext_vector_type(4)));
typedef unsigned int  u32x2 __attribute__((ext_vector_type(2)));

union BFrag  { v16bf bf; u32x4 q[2]; };
union Frag8  { v8i v; u32x4 q[2]; u32x2 d[4]; };

__device__ __forceinline__ u16 f2bf(float f) {
  union { float f; unsigned u; } x; x.f = f;
  unsigned u = x.u;
  unsigned r = u + 0x7FFFu + ((u >> 16) & 1u);
  if ((u & 0x7F800000u) == 0x7F800000u) r = u;
  return (u16)(r >> 16);
}

__device__ __forceinline__ u8 f2fp8_sw(float f) {
  unsigned u = __float_as_uint(f);
  u8 s = (u8)((u >> 24) & 0x80u);
  float a = __uint_as_float(u & 0x7FFFFFFFu);
  if (a >= 448.0f) return s | 0x7Eu;
  int e = (int)((u >> 23) & 0xFF);
  int exp8 = e - 127 + 7;
  if (exp8 <= 0) {
    int qi = (int)rintf(a * 512.0f);
    return s | (u8)qi;
  }
  int qi = (int)rintf(ldexpf(a, 10 - exp8));
  if (qi == 16) { qi = 8; exp8++; if (exp8 >= 16) return s | 0x7Eu; }
  return s | (u8)((exp8 << 3) | (qi & 7));
}

__device__ __forceinline__ u8 f2fp8(float f) {
#if defined(__has_builtin) && __has_builtin(__builtin_amdgcn_cvt_pk_fp8_f32)
  return (u8)(__builtin_amdgcn_cvt_pk_fp8_f32(f, 0.0f, 0, false) & 0xFF);
#else
  return f2fp8_sw(f);
#endif
}

__device__ __forceinline__ v16bf load_frag_a(const u16* p) {
  BFrag f;
  f.q[0] = *(const u32x4*)p;
  f.q[1] = *(const u32x4*)(p + 16);
  return f.bf;
}
__device__ __forceinline__ v16bf load_frag_b(const u16* p) {
  BFrag f;
  f.q[0] = *(const u32x4*)p;
  f.q[1] = *(const u32x4*)(p + 16);
  return f.bf;
}

__device__ __forceinline__ v8f wmma_bf16(v16bf a, v16bf b, v8f c) {
  return __builtin_amdgcn_wmma_f32_16x16x32_bf16(false, a, false, b, (short)0, c,
                                                 false, false);
}

__device__ __forceinline__ float fast_rcp(float x) { return __builtin_amdgcn_rcpf(x); }
__device__ __forceinline__ float sigmoidf(float x) {
  return fast_rcp(1.0f + __expf(-x));
}
__device__ __forceinline__ float tanh_fast(float x) {
  x = fminf(fmaxf(x, -30.0f), 30.0f);
  float e = __expf(-2.0f * x);
  return (1.0f - e) * fast_rcp(1.0f + e);
}

__global__ void cvt_f32_bf16_kernel(const float* __restrict__ src,
                                    u16* __restrict__ dst, int n) {
  int stride = gridDim.x * blockDim.x;
  for (int i = blockIdx.x * blockDim.x + threadIdx.x; i < n; i += stride)
    dst[i] = f2bf(src[i]);
}

__global__ void cvt_f32_fp8_kernel(const float* __restrict__ src,
                                   u8* __restrict__ dst, float scale, int n) {
  int stride = gridDim.x * blockDim.x;
  for (int i = blockIdx.x * blockDim.x + threadIdx.x; i < n; i += stride)
    dst[i] = f2fp8_sw(src[i] * scale);
}

__global__ void bias_sum_kernel(const float* __restrict__ a,
                                const float* __restrict__ b,
                                float* __restrict__ o, int n) {
  int i = blockIdx.x * blockDim.x + threadIdx.x;
  if (i < n) o[i] = a[i] + b[i];
}

__global__ __launch_bounds__(256) void gemm_xg_kernel(
    const u16* __restrict__ A, const u16* __restrict__ W,
    const float* __restrict__ bsum, float* __restrict__ out) {
  const int tid  = threadIdx.x;
  const int wave = tid >> 5;
  const int lane = tid & 31;
  const int l16  = lane & 15;
  const int half = lane >> 4;
  const size_t row0 = (size_t)blockIdx.y * 128 + (size_t)wave * 16;
  const int    n0   = blockIdx.x * 64;

  BFrag a[8];
#pragma unroll
  for (int kk = 0; kk < 8; ++kk) {
    const u16* ap = A + (row0 + l16) * 256 + kk * 32 + half * 8;
    a[kk].q[0] = *(const u32x4*)ap;
    a[kk].q[1] = *(const u32x4*)(ap + 16);
  }

  const u16* wbase[4];
#pragma unroll
  for (int j = 0; j < 4; ++j)
    wbase[j] = W + (size_t)(n0 + j * 16 + l16) * 256 + half * 8;

  v8f acc[4];
  BFrag b[4];
#pragma unroll
  for (int j = 0; j < 4; ++j) {
    acc[j] = (v8f){};
    b[j].q[0] = *(const u32x4*)(wbase[j]);
    b[j].q[1] = *(const u32x4*)(wbase[j] + 16);
  }

#pragma unroll
  for (int kk = 0; kk < 8; ++kk) {
    BFrag cur[4];
#pragma unroll
    for (int j = 0; j < 4; ++j) cur[j] = b[j];
    if (kk < 7) {
#pragma unroll
      for (int j = 0; j < 4; ++j) {
        const u16* p = wbase[j] + (kk + 1) * 32;
        b[j].q[0] = *(const u32x4*)p;
        b[j].q[1] = *(const u32x4*)(p + 16);
      }
    }
#pragma unroll
    for (int j = 0; j < 4; ++j)
      acc[j] = wmma_bf16(a[kk].bf, cur[j].bf, acc[j]);
  }

#pragma unroll
  for (int j = 0; j < 4; ++j) {
    const int col = n0 + j * 16 + l16;
    const float bb = bsum[col];
#pragma unroll
    for (int r = 0; r < 8; ++r)
      out[(row0 + half * 8 + r) * 1024 + col] = acc[j][r] + bb;
  }
}

#define HP 264
#define WP 272
__global__ __launch_bounds__(512) void lstm_scan_kernel(
    const float* __restrict__ xg,
    const u8*    __restrict__ whh8,
    u16*         __restrict__ hseq,
    u16*         __restrict__ hlast,
    int T) {
  __shared__ __align__(16) u8 wlds[1024 * WP];
  __shared__ __align__(16) u8 hbuf[16 * HP];

  const int tid  = threadIdx.x;
  const int wave = tid >> 5;
  const int lane = tid & 31;
  const int l16  = lane & 15;
  const int half = lane >> 4;
  const int b0   = blockIdx.x << 4;
  const int ncol = (wave << 4) + l16;

  for (int idx = tid; idx < 1024 * 16; idx += 512) {
    const int row = idx >> 4, ch = (idx & 15) * 16;
    *(u32x4*)&wlds[row * WP + ch] = *(const u32x4*)&whh8[row * 256 + ch];
  }
  for (int i = tid; i < 16 * HP; i += 512) hbuf[i] = 0;

  float c[8];
#pragma unroll
  for (int r = 0; r < 8; ++r) c[r] = 0.0f;
  float hv[8];
#pragma unroll
  for (int r = 0; r < 8; ++r) hv[r] = 0.0f;

  const u8* wr[4];
#pragma unroll
  for (int cc = 0; cc < 4; ++cc) wr[cc] = &wlds[(size_t)(cc * 256 + ncol) * WP];

  __syncthreads();

  const float descale = 1.0f / 1024.0f;

  for (int t = 0; t < T; ++t) {
    if (t + 1 < T) {
      const int pb = tid >> 5, seg = (tid & 31) * 32;
      const float* pf = xg + ((size_t)(b0 + pb) * T + (t + 1)) * 1024 + seg;
      __builtin_prefetch((const void*)pf, 0, 0);
    }

    v8f acc[4];
#pragma unroll
    for (int cc = 0; cc < 4; ++cc) acc[cc] = (v8f){};

#pragma unroll
    for (int ks = 0; ks < 4; ++ks) {
      const u8* ap = &hbuf[l16 * HP + ks * 64 + half * 16];
      Frag8 a;
      a.d[0] = *(const u32x2*)(ap);
      a.d[1] = *(const u32x2*)(ap + 8);
      a.d[2] = *(const u32x2*)(ap + 32);
      a.d[3] = *(const u32x2*)(ap + 40);
#pragma unroll
      for (int cc = 0; cc < 4; ++cc) {
        const u8* bp = wr[cc] + ks * 64 + half * 16;
        Frag8 b;
        b.q[0] = *(const u32x4*)(bp);
        b.q[1] = *(const u32x4*)(bp + 32);
        acc[cc] = __builtin_amdgcn_wmma_f32_16x16x64_fp8_fp8(
            a.v, b.v, (short)0, acc[cc], false, false);
      }
    }

#pragma unroll
    for (int r = 0; r < 8; ++r) {
      const int m = half * 8 + r;
      const size_t base = ((size_t)(b0 + m) * T + t) * 1024;
      float gi = acc[0][r] * descale + xg[base + 0 * 256 + ncol];
      float gf = acc[1][r] * descale + xg[base + 1 * 256 + ncol];
      float gg = acc[2][r] * descale + xg[base + 2 * 256 + ncol];
      float go = acc[3][r] * descale + xg[base + 3 * 256 + ncol];
      c[r]  = sigmoidf(gf) * c[r] + sigmoidf(gi) * tanh_fast(gg);
      hv[r] = sigmoidf(go) * tanh_fast(c[r]);
    }

    __syncthreads();
#pragma unroll
    for (int r = 0; r < 8; ++r) {
      const int m = half * 8 + r;
      hbuf[m * HP + ncol] = f2fp8(hv[r] * 16.0f);
      if (hseq) hseq[((size_t)(b0 + m) * T + t) * 256 + ncol] = f2bf(hv[r]);
    }
    __syncthreads();
  }

  if (hlast) {
#pragma unroll
    for (int r = 0; r < 8; ++r)
      hlast[(size_t)(b0 + half * 8 + r) * 256 + ncol] = f2bf(hv[r]);
  }
}

__global__ __launch_bounds__(256) void fc_kernel(
    const u16* __restrict__ h, const u16* __restrict__ W,
    const float* __restrict__ bias, float* __restrict__ out) {
  const int tid  = threadIdx.x;
  const int wave = tid >> 5;
  const int lane = tid & 31;
  const int l16  = lane & 15;
  const int half = lane >> 4;

#pragma unroll
  for (int i = 0; i < 4; ++i) {
    const int tile = wave * 4 + i;
    const int mt = tile & 1;
    const int nt = tile >> 1;
    v8f acc = (v8f){};
#pragma unroll
    for (int kk = 0; kk < 8; ++kk) {
      v16bf a = load_frag_a(h + (size_t)(mt * 16 + l16) * 256 + kk * 32 + half * 8);
      v16bf b = load_frag_b(W + (size_t)(nt * 16 + l16) * 256 + kk * 32 + half * 8);
      acc = wmma_bf16(a, b, acc);
    }
    const int col = nt * 16 + l16;
    const float bb = bias[col];
#pragma unroll
    for (int r = 0; r < 8; ++r)
      out[(size_t)(mt * 16 + half * 8 + r) * 256 + col] = acc[r] + bb;
  }
}

extern "C" void kernel_launch(void* const* d_in, const int* in_sizes, int n_in,
                              void* d_out, int out_size, void* d_ws, size_t ws_size,
                              hipStream_t stream) {
  (void)n_in; (void)out_size; (void)ws_size;

  const float* x    = (const float*)d_in[0];
  const float* Wih0 = (const float*)d_in[1];
  const float* Whh0 = (const float*)d_in[2];
  const float* bih0 = (const float*)d_in[3];
  const float* bhh0 = (const float*)d_in[4];
  const float* Wih1 = (const float*)d_in[5];
  const float* Whh1 = (const float*)d_in[6];
  const float* bih1 = (const float*)d_in[7];
  const float* bhh1 = (const float*)d_in[8];
  const float* Wfc  = (const float*)d_in[9];
  const float* bfc  = (const float*)d_in[10];

  const int B = 32, D = 256, H = 256, G = 4 * H;
  const int T = in_sizes[0] / (B * D);
  const size_t MT = (size_t)B * T;

  char* ws = (char*)d_ws;
  size_t off = 0;
  auto carve = [&](size_t bytes) -> char* {
    char* p = ws + off;
    off += (bytes + 255) & ~(size_t)255;
    return p;
  };
  u16*   xbf    = (u16*)  carve(MT * D * sizeof(u16));
  u16*   h0bf   = (u16*)  carve(MT * H * sizeof(u16));
  float* xgbuf  = (float*)carve(MT * G * sizeof(float));
  u16*   wih0b  = (u16*)  carve((size_t)G * D * sizeof(u16));
  u16*   wih1b  = (u16*)  carve((size_t)G * H * sizeof(u16));
  u16*   wfcb   = (u16*)  carve((size_t)D * H * sizeof(u16));
  u8*    whh0f8 = (u8*)   carve((size_t)G * H);
  u8*    whh1f8 = (u8*)   carve((size_t)G * H);
  float* bsum0  = (float*)carve(G * sizeof(float));
  float* bsum1  = (float*)carve(G * sizeof(float));
  u16*   hlastb = (u16*)  carve((size_t)B * H * sizeof(u16));

  auto cvt = [&](const float* s, u16* d, size_t n) {
    int blocks = (int)((n + 255) / 256);
    if (blocks > 4096) blocks = 4096;
    cvt_f32_bf16_kernel<<<blocks, 256, 0, stream>>>(s, d, (int)n);
  };

  cvt(x,    xbf,   MT * D);
  cvt(Wih0, wih0b, (size_t)G * D);
  cvt(Wih1, wih1b, (size_t)G * H);
  cvt(Wfc,  wfcb,  (size_t)D * H);
  cvt_f32_fp8_kernel<<<1024, 256, 0, stream>>>(Whh0, whh0f8, 64.0f, G * H);
  cvt_f32_fp8_kernel<<<1024, 256, 0, stream>>>(Whh1, whh1f8, 64.0f, G * H);

  bias_sum_kernel<<<4, 256, 0, stream>>>(bih0, bhh0, bsum0, G);
  bias_sum_kernel<<<4, 256, 0, stream>>>(bih1, bhh1, bsum1, G);

  dim3 gemm_grid(G / 64, (unsigned)(MT / 128));

  gemm_xg_kernel<<<gemm_grid, 256, 0, stream>>>(xbf, wih0b, bsum0, xgbuf);
  lstm_scan_kernel<<<2, 512, 0, stream>>>(xgbuf, whh0f8, h0bf, nullptr, T);
  gemm_xg_kernel<<<gemm_grid, 256, 0, stream>>>(h0bf, wih1b, bsum1, xgbuf);
  lstm_scan_kernel<<<2, 512, 0, stream>>>(xgbuf, whh1f8, nullptr, hlastb, T);
  fc_kernel<<<1, 256, 0, stream>>>(hlastb, wfcb, bfc, (float*)d_out);
}
